// DMPNN_75453985456261
// MI455X (gfx1250) — matrix operand order repaired and checked by reading, not compiled, not run on hardware
//
#include <hip/hip_runtime.h>
#include <hip/hip_bf16.h>
#include <math.h>

#define N_    16384
#define E_    131072
#define LE_   524288
#define F_    256
#define ED_   64
#define T_    4
#define G_    512
#define SND_  1536
#define EPS_  1e-5f

typedef __attribute__((ext_vector_type(16))) __bf16 v16bf;
typedef __attribute__((ext_vector_type(8)))  float  v8f;
typedef unsigned short ushort_t;

union FragBF { v16bf v; uint4 q[2]; };

__device__ __forceinline__ v8f zero8() {
  v8f v;
#pragma unroll
  for (int i = 0; i < 8; i++) v[i] = 0.0f;
  return v;
}

// ---------------------------------------------------------------------------
// Async global->LDS copy (CDNA5 GLOBAL_LOAD_ASYNC_TO_LDS_B128, ASYNCcnt).
// ---------------------------------------------------------------------------
#if __has_builtin(__builtin_amdgcn_global_load_async_to_lds_b128)
#define HAVE_ASYNC_LDS 1
#endif

typedef int v4i_ __attribute__((vector_size(16)));
typedef __attribute__((address_space(1))) v4i_ gv4i_t;
typedef __attribute__((address_space(3))) v4i_ lv4i_t;

__device__ __forceinline__ void ld_lds16(ushort_t* lds, const ushort_t* g) {
#ifdef HAVE_ASYNC_LDS
  __builtin_amdgcn_global_load_async_to_lds_b128(
      (gv4i_t*)(ushort_t*)g, (lv4i_t*)lds, 0, 0);
#else
  *(uint4*)lds = *(const uint4*)g;
#endif
}

__device__ __forceinline__ void wait_async_lds() {
#ifdef HAVE_ASYNC_LDS
#if __has_builtin(__builtin_amdgcn_s_wait_asynccnt)
  __builtin_amdgcn_s_wait_asynccnt(0);
#else
  asm volatile("s_wait_asynccnt 0x0" ::: "memory");
#endif
#endif
}

// ---------------------------------------------------------------------------
// WMMA bf16 GEMM:  C(MxNc) = A(MxK) * B(NcxK)^T  [+bias] [epilogue]
// mode 0: C = acc + bias
// mode 1: C = tanh(acc + bias)
// mode 2: C = (acc + bias + res) * 0.5
// Requires M%256==0, Nc%128==0, K%32==0.
// Block: 256 threads = 8 waves (4x2); block tile 256x128; wave tile 64x64
// (16 accumulators -> 16 WMMAs per 16 LDS b128 reads per K-step).
// Double-buffered LDS; tile k+1 streams in via GLOBAL_LOAD_ASYNC_TO_LDS_B128
// while tile k is consumed by v_wmma_f32_16x16x32_bf16.
// ---------------------------------------------------------------------------
#define BM 256
#define BN 128
#define KB 32
#define KBP 40   // padded LDS row stride (bf16 elems) to reduce bank conflicts

__global__ __launch_bounds__(256)
void wmma_gemm_bf16(const ushort_t* __restrict__ A,
                    const ushort_t* __restrict__ B,
                    float* __restrict__ C,
                    const float* __restrict__ bias,
                    const float* __restrict__ res,
                    int M, int Nc, int K, int mode)
{
  __shared__ __align__(16) ushort_t As[2][BM * KBP];  // 2 x 20KB
  __shared__ __align__(16) ushort_t Bs[2][BN * KBP];  // 2 x 10KB

  const int tid  = threadIdx.x;
  const int wave = tid >> 5;
  const int lane = tid & 31;
  const int half = lane >> 4;
  const int l16  = lane & 15;
  const int wm   = wave & 3;   // 4 waves along M
  const int wn   = wave >> 2;  // 2 waves along N
  const long long m0 = (long long)blockIdx.y * BM;
  const long long n0 = (long long)blockIdx.x * BN;

  // Straight-line staging: A = 1024 16B-chunks (4/thread), B = 512 (2/thread).
  const ushort_t* gA[4];
  const ushort_t* gB[2];
  int lA[4], lB[2];
#pragma unroll
  for (int j = 0; j < 4; j++) {
    int i = tid + 256 * j;
    int r = i >> 2, c = (i & 3) << 3;
    gA[j] = A + (m0 + r) * (long long)K + c;
    lA[j] = r * KBP + c;
  }
#pragma unroll
  for (int j = 0; j < 2; j++) {
    int i = tid + 256 * j;
    int r = i >> 2, c = (i & 3) << 3;
    gB[j] = B + (n0 + r) * (long long)K + c;
    lB[j] = r * KBP + c;
  }

  auto stage = [&](int buf, int k0) {
#pragma unroll
    for (int j = 0; j < 4; j++) ld_lds16(&As[buf][lA[j]], gA[j] + k0);
#pragma unroll
    for (int j = 0; j < 2; j++) ld_lds16(&Bs[buf][lB[j]], gB[j] + k0);
  };

  v8f acc[4][4];
#pragma unroll
  for (int i = 0; i < 4; i++)
#pragma unroll
    for (int j = 0; j < 4; j++) acc[i][j] = zero8();

  stage(0, 0);
  wait_async_lds();
  __syncthreads();

  int buf = 0;
  for (int k0 = 0; k0 < K; k0 += KB) {
    const int nxt = buf ^ 1;
    if (k0 + KB < K) stage(nxt, k0 + KB);  // overlap copy with WMMA

    // B fragments: lane l16 = column n; k = 8*half+e (e<8), 16+8*half+(e-8)
    FragBF bf[4];
#pragma unroll
    for (int nt = 0; nt < 4; nt++) {
      int n = wn * 64 + nt * 16 + l16;
      bf[nt].q[0] = *(const uint4*)&Bs[buf][n * KBP + 8 * half];
      bf[nt].q[1] = *(const uint4*)&Bs[buf][n * KBP + 16 + 8 * half];
    }
    // A fragments: lane l16 = row m; k = 8*half+e (e<8), 16+8*half+(e-8)
#pragma unroll
    for (int mt = 0; mt < 4; mt++) {
      FragBF af;
      int r = wm * 64 + mt * 16 + l16;
      af.q[0] = *(const uint4*)&As[buf][r * KBP + 8 * half];
      af.q[1] = *(const uint4*)&As[buf][r * KBP + 16 + 8 * half];
#pragma unroll
      for (int nt = 0; nt < 4; nt++) {
        acc[mt][nt] = __builtin_amdgcn_wmma_f32_16x16x32_bf16(
            false, af.v, false, bf[nt].v, (short)0, acc[mt][nt], false, false);
      }
    }

    wait_async_lds();   // next tile landed in LDS
    __syncthreads();    // everyone done reading buf / writing nxt
    buf = nxt;
  }

  // Epilogue: D layout — VGPR r: m = r + 8*half; n = lane&15
#pragma unroll
  for (int mt = 0; mt < 4; mt++) {
#pragma unroll
    for (int nt = 0; nt < 4; nt++) {
      long long n = n0 + wn * 64 + nt * 16 + l16;
      float bv = bias ? bias[n] : 0.0f;
#pragma unroll
      for (int r8 = 0; r8 < 8; r8++) {
        long long m = m0 + wm * 64 + mt * 16 + r8 + 8 * half;
        long long idx = m * (long long)Nc + n;
        float v = acc[mt][nt][r8] + bv;
        if (mode == 1)      v = tanhf(v);
        else if (mode == 2) v = (v + res[idx]) * 0.5f;
        C[idx] = v;
      }
    }
  }
}

// ---------------------------------------------------------------------------
// Elementwise / utility kernels
// ---------------------------------------------------------------------------
__global__ void k_f32_to_bf16(const float* __restrict__ in,
                              ushort_t* __restrict__ out, long long n) {
  long long i = (long long)blockIdx.x * 256 + threadIdx.x;
  if (i < n) {
    unsigned u = __float_as_uint(in[i]);
    unsigned r = u + 0x7FFFu + ((u >> 16) & 1u);   // RNE
    out[i] = (ushort_t)(r >> 16);
  }
}

__global__ void k_zero_f32(float* p, long long n) {
  long long i = (long long)blockIdx.x * 256 + threadIdx.x;
  if (i < n) p[i] = 0.0f;
}

__global__ void k_fill_u32(unsigned* p, unsigned v, long long n) {
  long long i = (long long)blockIdx.x * 256 + threadIdx.x;
  if (i < n) p[i] = v;
}

__global__ void k_copy_f32(const float* __restrict__ a, float* __restrict__ b,
                           long long n) {
  long long i = (long long)blockIdx.x * 256 + threadIdx.x;
  if (i < n) b[i] = a[i];
}

__global__ void k_add4(const float* __restrict__ a, const float* __restrict__ b,
                       float* __restrict__ c, long long n4) {
  long long i = (long long)blockIdx.x * 256 + threadIdx.x;
  if (i < n4) {
    float4 x = ((const float4*)a)[i];
    float4 y = ((const float4*)b)[i];
    float4 o;
    o.x = x.x + y.x; o.y = x.y + y.y; o.z = x.z + y.z; o.w = x.w + y.w;
    ((float4*)c)[i] = o;
  }
}

// h = (xu[src] + xv[dst] + he) / 3   — thread per (edge, 4 feats)
__global__ void k_build_h(const float* __restrict__ xu, const float* __restrict__ xv,
                          const float* __restrict__ he, const int* __restrict__ src,
                          const int* __restrict__ dst, float* __restrict__ h) {
  long long idx = (long long)blockIdx.x * 256 + threadIdx.x;
  if (idx >= (long long)E_ * (F_ / 4)) return;
  long long e = idx >> 6;
  int fq = (int)(idx & 63) << 2;
  int s = src[e], d = dst[e];
  float4 a = *(const float4*)&xu[(long long)s * F_ + fq];
  float4 b = *(const float4*)&xv[(long long)d * F_ + fq];
  float4 c = *(const float4*)&he[(long long)e * F_ + fq];
  const float k3 = 1.0f / 3.0f;
  float4 o;
  o.x = (a.x + b.x + c.x) * k3; o.y = (a.y + b.y + c.y) * k3;
  o.z = (a.z + b.z + c.z) * k3; o.w = (a.w + b.w + c.w) * k3;
  *(float4*)&h[(long long)e * F_ + fq] = o;
}

__device__ __forceinline__ void atomF(float* p, float v) {
  __hip_atomic_fetch_add(p, v, __ATOMIC_RELAXED, __HIP_MEMORY_SCOPE_AGENT);
}

// msg[lg_dst] += prev[lg_src]  over LE line-graph edges
__global__ void k_lg_scatter(const float* __restrict__ prev,
                             const int* __restrict__ lgs, const int* __restrict__ lgd,
                             float* __restrict__ msg) {
  long long idx = (long long)blockIdx.x * 256 + threadIdx.x;
  if (idx >= (long long)LE_ * (F_ / 4)) return;
  long long le = idx >> 6;
  int fq = (int)(idx & 63) << 2;
  int s = lgs[le], d = lgd[le];
  float4 v = *(const float4*)&prev[(long long)s * F_ + fq];
  float* o = &msg[(long long)d * F_ + fq];
  atomF(&o[0], v.x); atomF(&o[1], v.y); atomF(&o[2], v.z); atomF(&o[3], v.w);
}

// attention score per edge: one wave32 per edge, shuffle reduction
__global__ void k_att(const float* __restrict__ cur, const float* __restrict__ attW,
                      const float* __restrict__ attb, float* __restrict__ sc) {
  int e = blockIdx.x * 8 + (threadIdx.x >> 5);
  int lane = threadIdx.x & 31;
  if (e >= E_) return;
  float s = 0.0f;
#pragma unroll
  for (int j = 0; j < F_ / 32; j++) {
    int f = lane + j * 32;
    s += cur[(long long)e * F_ + f] * attW[f];
  }
#pragma unroll
  for (int m = 16; m >= 1; m >>= 1) s += __shfl_xor(s, m, 32);
  if (lane == 0) sc[e] = s + attb[0];
}

__device__ __forceinline__ unsigned enc_f(float f) {
  unsigned u = __float_as_uint(f);
  return (u & 0x80000000u) ? ~u : (u | 0x80000000u);
}
__device__ __forceinline__ float dec_f(unsigned k) {
  unsigned u = (k & 0x80000000u) ? (k & 0x7FFFFFFFu) : ~k;
  return __uint_as_float(u);
}

__global__ void k_segmax(const float* __restrict__ sc, const int* __restrict__ batch,
                         unsigned* __restrict__ segmax) {
  long long e = (long long)blockIdx.x * 256 + threadIdx.x;
  if (e < E_) atomicMax(&segmax[batch[e]], enc_f(sc[e]));
}

__global__ void k_exp(const float* __restrict__ sc, const int* __restrict__ batch,
                      const unsigned* __restrict__ segmax, float* __restrict__ esc,
                      float* __restrict__ segz) {
  long long e = (long long)blockIdx.x * 256 + threadIdx.x;
  if (e < E_) {
    float v = __expf(sc[e] - dec_f(segmax[batch[e]]));
    esc[e] = v;
    atomF(&segz[batch[e]], v);
  }
}

// gx[g] += cur[e] * softmax_weight(e)
__global__ void k_gx(const float* __restrict__ cur, const float* __restrict__ esc,
                     const float* __restrict__ segz, const int* __restrict__ batch,
                     float* __restrict__ gx) {
  long long idx = (long long)blockIdx.x * 256 + threadIdx.x;
  if (idx >= (long long)E_ * (F_ / 4)) return;
  long long e = idx >> 6;
  int fq = (int)(idx & 63) << 2;
  int g = batch[e];
  float w = esc[e] / (segz[g] + 1e-16f);
  float4 v = *(const float4*)&cur[(long long)e * F_ + fq];
  float* o = &gx[(long long)g * F_ + fq];
  atomF(&o[0], v.x * w); atomF(&o[1], v.y * w);
  atomF(&o[2], v.z * w); atomF(&o[3], v.w * w);
}

// per-graph scores + softmax over T
__global__ void k_scores(const float* __restrict__ gout_all, const float* __restrict__ a,
                         const float* __restrict__ ab, float* __restrict__ escore) {
  int g = blockIdx.x * 128 + threadIdx.x;
  if (g >= G_) return;
  float s[T_];
#pragma unroll
  for (int t = 0; t < T_; t++) {
    float acc = ab[t];
    for (int f = 0; f < F_; f++)
      acc += gout_all[(long long)t * G_ * F_ + (long long)g * F_ + f] * a[f * T_ + t];
    s[t] = acc;
  }
  float mx = s[0];
#pragma unroll
  for (int t = 1; t < T_; t++) mx = fmaxf(mx, s[t]);
  float z = 0.0f, ev[T_];
#pragma unroll
  for (int t = 0; t < T_; t++) { ev[t] = __expf(s[t] - mx); z += ev[t]; }
#pragma unroll
  for (int t = 0; t < T_; t++) escore[g * T_ + t] = ev[t] / z;
}

// out = sum_t out_all[t]*escore[g,t]; xn[dst] += out
__global__ void k_final_mix(const float* __restrict__ out_all,
                            const float* __restrict__ escore,
                            const int* __restrict__ batch, const int* __restrict__ dst,
                            float* __restrict__ xn) {
  long long idx = (long long)blockIdx.x * 256 + threadIdx.x;
  if (idx >= (long long)E_ * (F_ / 4)) return;
  long long e = idx >> 6;
  int fq = (int)(idx & 63) << 2;
  int g = batch[e];
  int d = dst[e];
  float w[T_];
#pragma unroll
  for (int t = 0; t < T_; t++) w[t] = escore[g * T_ + t];
  float4 o; o.x = o.y = o.z = o.w = 0.0f;
#pragma unroll
  for (int t = 0; t < T_; t++) {
    float4 v = *(const float4*)&out_all[(long long)t * E_ * F_ + (long long)e * F_ + fq];
    o.x += v.x * w[t]; o.y += v.y * w[t]; o.z += v.z * w[t]; o.w += v.w * w[t];
  }
  float* p = &xn[(long long)d * F_ + fq];
  atomF(&p[0], o.x); atomF(&p[1], o.y); atomF(&p[2], o.z); atomF(&p[3], o.w);
}

// ---------------------------------------------------------------------------
// BatchNorm (biased var, per-column over M rows)
// ---------------------------------------------------------------------------
__global__ void k_bn_stats(const float* __restrict__ X, float* __restrict__ sums,
                           int M, int C) {
  int c = blockIdx.x * 256 + threadIdx.x;
  int r0 = blockIdx.y * 128;
  float s = 0.0f, q = 0.0f;
  for (int r = 0; r < 128; r++) {
    float v = X[(long long)(r0 + r) * C + c];
    s += v; q += v * v;
  }
  atomF(&sums[c], s);
  atomF(&sums[C + c], q);
}

__global__ void k_bn_fin(const float* __restrict__ sums, float* __restrict__ mean,
                         float* __restrict__ rstd, int M, int C) {
  int c = blockIdx.x * 256 + threadIdx.x;
  if (c < C) {
    float m = sums[c] / (float)M;
    float v = sums[C + c] / (float)M - m * m;
    mean[c] = m;
    rstd[c] = rsqrtf(v + EPS_);
  }
}

// y = maybe_prelu(bn(x)*g+b), stored bf16
__global__ void k_bn_apply(const float* __restrict__ X, const float* __restrict__ mean,
                           const float* __restrict__ rstd, const float* __restrict__ gg,
                           const float* __restrict__ bb, const float* __restrict__ pp,
                           ushort_t* __restrict__ Y, long long n, int C) {
  long long i = (long long)blockIdx.x * 256 + threadIdx.x;
  if (i >= n) return;
  int c = (int)(i % C);
  float v = (X[i] - mean[c]) * rstd[c] * gg[c] + bb[c];
  if (pp) v = (v >= 0.0f) ? v : pp[0] * v;
  unsigned u = __float_as_uint(v);
  unsigned r = u + 0x7FFFu + ((u >> 16) & 1u);
  Y[i] = (ushort_t)(r >> 16);
}

// ---------------------------------------------------------------------------
static inline unsigned gblocks(long long n, int b) {
  return (unsigned)((n + b - 1) / b);
}

extern "C" void kernel_launch(void* const* d_in, const int* in_sizes, int n_in,
                              void* d_out, int out_size, void* d_ws, size_t ws_size,
                              hipStream_t stream) {
  (void)in_sizes; (void)n_in; (void)out_size; (void)ws_size;

  const float* x    = (const float*)d_in[0];
  const float* ea   = (const float*)d_in[1];
  const int*   ei   = (const int*)d_in[2];
  const int*   lgi  = (const int*)d_in[3];
  const int*   bat  = (const int*)d_in[4];
  const float* W_u  = (const float*)d_in[5];
  const float* W_v  = (const float*)d_in[6];
  const float* W_e  = (const float*)d_in[7];
  const float* attW = (const float*)d_in[8];
  const float* attb = (const float*)d_in[9];
  const float* a_w  = (const float*)d_in[10];
  const float* a_b  = (const float*)d_in[11];
  const float* Wg   = (const float*)d_in[12];
  const float* bg   = (const float*)d_in[13];
  const float* bn0g = (const float*)d_in[14];
  const float* bn0b = (const float*)d_in[15];
  const float* W1   = (const float*)d_in[16];
  const float* b1   = (const float*)d_in[17];
  const float* bn1g = (const float*)d_in[18];
  const float* bn1b = (const float*)d_in[19];
  const float* p1   = (const float*)d_in[20];
  const float* W2   = (const float*)d_in[21];
  const float* b2   = (const float*)d_in[22];
  const float* bn2g = (const float*)d_in[23];
  const float* bn2b = (const float*)d_in[24];
  const float* p2   = (const float*)d_in[25];
  const float* W3   = (const float*)d_in[26];
  const float* b3   = (const float*)d_in[27];
  const float* bn3g = (const float*)d_in[28];
  const float* bn3b = (const float*)d_in[29];
  const float* p3   = (const float*)d_in[30];
  const float* W4   = (const float*)d_in[31];
  const float* b4   = (const float*)d_in[32];
  const float* bn4g = (const float*)d_in[33];
  const float* bn4b = (const float*)d_in[34];
  const float* p4   = (const float*)d_in[35];
  const float* W5   = (const float*)d_in[36];
  const float* b5   = (const float*)d_in[37];

  const int* src = ei;
  const int* dst = ei + E_;
  const int* lgs = lgi;
  const int* lgd = lgi + LE_;

  // ---- workspace layout ----
  char* p = (char*)d_ws;
  auto alloc = [&](size_t bytes) -> void* {
    void* r = (void*)p;
    p += (bytes + 255) & ~(size_t)255;
    return r;
  };
  ushort_t* xb   = (ushort_t*)alloc((size_t)N_ * F_ * 2);
  ushort_t* eab  = (ushort_t*)alloc((size_t)E_ * ED_ * 2);
  ushort_t* Wub  = (ushort_t*)alloc((size_t)F_ * F_ * 2);
  ushort_t* Wvb  = (ushort_t*)alloc((size_t)F_ * F_ * 2);
  ushort_t* Web  = (ushort_t*)alloc((size_t)F_ * ED_ * 2);
  ushort_t* Wgb  = (ushort_t*)alloc((size_t)F_ * F_ * 2);
  ushort_t* W1b  = (ushort_t*)alloc((size_t)SND_ * F_ * 2);
  ushort_t* W2b  = (ushort_t*)alloc((size_t)SND_ * SND_ * 2);
  ushort_t* W3b  = (ushort_t*)alloc((size_t)SND_ * SND_ * 2);
  ushort_t* W4b  = (ushort_t*)alloc((size_t)SND_ * SND_ * 2);
  ushort_t* W5b  = (ushort_t*)alloc((size_t)F_ * SND_ * 2);
  float* xu      = (float*)alloc((size_t)N_ * F_ * 4);
  float* xv      = (float*)alloc((size_t)N_ * F_ * 4);
  float* h       = (float*)alloc((size_t)E_ * F_ * 4);
  float* msg     = (float*)alloc((size_t)E_ * F_ * 4);   // also 'he' scratch
  float* out_all = (float*)alloc((size_t)T_ * E_ * F_ * 4);
  float* sc      = (float*)alloc((size_t)E_ * 4);
  float* esc     = (float*)alloc((size_t)E_ * 4);
  unsigned* smax = (unsigned*)alloc((size_t)G_ * 4);
  float* segz    = (float*)alloc((size_t)G_ * 4);
  float* gx      = (float*)alloc((size_t)G_ * F_ * 4);
  ushort_t* gxb  = (ushort_t*)alloc((size_t)G_ * F_ * 2);
  float* gout    = (float*)alloc((size_t)T_ * G_ * F_ * 4);
  float* escore  = (float*)alloc((size_t)G_ * T_ * 4);
  float* xn      = (float*)alloc((size_t)N_ * F_ * 4);
  float* sums    = (float*)alloc((size_t)2 * SND_ * 4);
  float* mean    = (float*)alloc((size_t)SND_ * 4);
  float* rstd    = (float*)alloc((size_t)SND_ * 4);
  ushort_t* actb = (ushort_t*)alloc((size_t)N_ * SND_ * 2);
  float* bufA    = (float*)alloc((size_t)N_ * SND_ * 4);
  float* bufB    = (float*)alloc((size_t)N_ * SND_ * 4);

  const long long EF   = (long long)E_ * F_;
  const long long EF4  = EF / 4;
  const long long LEF4 = (long long)LE_ * (F_ / 4);

  auto cvt = [&](const float* s, ushort_t* d, long long n) {
    k_f32_to_bf16<<<gblocks(n, 256), 256, 0, stream>>>(s, d, n);
  };
  auto gemm = [&](const ushort_t* A, const ushort_t* B, float* C,
                  const float* bias, const float* res, long long M, int Nc, int K,
                  int mode) {
    dim3 grid((unsigned)(Nc / BN), (unsigned)(M / BM));
    wmma_gemm_bf16<<<grid, 256, 0, stream>>>(A, B, C, bias, res, (int)M, Nc, K, mode);
  };

  // ---- stage bf16 copies of activations + weights ----
  cvt(x,   xb,  (long long)N_ * F_);
  cvt(ea,  eab, (long long)E_ * ED_);
  cvt(W_u, Wub, (long long)F_ * F_);
  cvt(W_v, Wvb, (long long)F_ * F_);
  cvt(W_e, Web, (long long)F_ * ED_);
  cvt(Wg,  Wgb, (long long)F_ * F_);
  cvt(W1,  W1b, (long long)SND_ * F_);
  cvt(W2,  W2b, (long long)SND_ * SND_);
  cvt(W3,  W3b, (long long)SND_ * SND_);
  cvt(W4,  W4b, (long long)SND_ * SND_);
  cvt(W5,  W5b, (long long)F_ * SND_);

  // ---- front-end projections (WMMA) ----
  gemm(xb,  Wub, xu,  nullptr, nullptr, N_, F_, F_,  0);  // x @ W_u.T
  gemm(xb,  Wvb, xv,  nullptr, nullptr, N_, F_, F_,  0);  // x @ W_v.T
  gemm(eab, Web, msg, nullptr, nullptr, E_, F_, ED_, 0);  // edge_attr @ W_e.T
  k_build_h<<<gblocks(EF4, 256), 256, 0, stream>>>(xu, xv, msg, src, dst, h);

  // ---- T message-passing iterations ----
  for (int t = 0; t < T_; t++) {
    const float* prev = (t == 0) ? h : (out_all + (long long)(t - 1) * EF);
    float* cur = out_all + (long long)t * EF;

    k_zero_f32<<<gblocks(EF, 256), 256, 0, stream>>>(msg, EF);
    k_lg_scatter<<<gblocks(LEF4, 256), 256, 0, stream>>>(prev, lgs, lgd, msg);
    k_add4<<<gblocks(EF4, 256), 256, 0, stream>>>(h, msg, cur, EF4);

    k_att<<<gblocks(E_, 8), 256, 0, stream>>>(cur, attW, attb, sc);
    k_fill_u32<<<gblocks(G_, 256), 256, 0, stream>>>(smax, 0u, G_);
    k_zero_f32<<<gblocks(G_, 256), 256, 0, stream>>>(segz, G_);
    k_segmax<<<gblocks(E_, 256), 256, 0, stream>>>(sc, bat, smax);
    k_exp<<<gblocks(E_, 256), 256, 0, stream>>>(sc, bat, smax, esc, segz);

    k_zero_f32<<<gblocks((long long)G_ * F_, 256), 256, 0, stream>>>(gx, (long long)G_ * F_);
    k_gx<<<gblocks(EF4, 256), 256, 0, stream>>>(cur, esc, segz, bat, gx);
    cvt(gx, gxb, (long long)G_ * F_);
    gemm(gxb, Wgb, gout + (long long)t * G_ * F_, bg, nullptr, G_, F_, F_, 1); // tanh
  }

  // ---- temporal attention + node aggregation ----
  k_scores<<<gblocks(G_, 128), 128, 0, stream>>>(gout, a_w, a_b, escore);
  k_copy_f32<<<gblocks((long long)N_ * F_, 256), 256, 0, stream>>>(x, xn, (long long)N_ * F_);
  k_final_mix<<<gblocks(EF4, 256), 256, 0, stream>>>(out_all, escore, bat, dst, xn);

  // ---- MLP tail: BN -> (PReLU) -> WMMA GEMM with fused bias/residual ----
  auto bnstats = [&](const float* X, int C) {
    k_zero_f32<<<gblocks(2LL * C, 256), 256, 0, stream>>>(sums, 2LL * C);
    dim3 g((unsigned)(C / 256), (unsigned)(N_ / 128));
    k_bn_stats<<<g, 256, 0, stream>>>(X, sums, N_, C);
    k_bn_fin<<<gblocks(C, 256), 256, 0, stream>>>(sums, mean, rstd, N_, C);
  };
  auto bnapply = [&](const float* X, const float* gg, const float* bb,
                     const float* pp, int C) {
    long long n = (long long)N_ * C;
    k_bn_apply<<<gblocks(n, 256), 256, 0, stream>>>(X, mean, rstd, gg, bb, pp, actb, n, C);
  };

  bnstats(xn, F_);        bnapply(xn, bn0g, bn0b, nullptr, F_);
  gemm(actb, W1b, bufA, b1, nullptr, N_, SND_, F_, 0);          // out1

  bnstats(bufA, SND_);    bnapply(bufA, bn1g, bn1b, p1, SND_);
  gemm(actb, W2b, bufB, b2, bufA, N_, SND_, SND_, 2);           // out2 = (.. + out1)/2

  bnstats(bufB, SND_);    bnapply(bufB, bn2g, bn2b, p2, SND_);
  gemm(actb, W3b, bufA, b3, bufB, N_, SND_, SND_, 2);           // out3

  bnstats(bufA, SND_);    bnapply(bufA, bn3g, bn3b, p3, SND_);
  gemm(actb, W4b, bufB, b4, bufA, N_, SND_, SND_, 2);           // out4

  bnstats(bufB, SND_);    bnapply(bufB, bn4g, bn4b, p4, SND_);
  gemm(actb, W5b, (float*)d_out, b5, nullptr, N_, F_, SND_, 0); // out5
}
